// DeformAtten2D_35356170780739
// MI455X (gfx1250) — matrix operand order repaired and checked by reading, not compiled, not run on hardware
//
#include <hip/hip_runtime.h>

typedef __attribute__((ext_vector_type(16))) __bf16 v16bf;
typedef __attribute__((ext_vector_type(8)))  float  v8f;

#define HW     128
#define NC     64
#define NBATCH 16
#define NPIX   (NBATCH * HW * HW)      /* 262144 pixels */
#define NBUF   ((size_t)NPIX * NC)     /* 16777216 elems per tensor */

// ---------------------------------------------------------------------------
// WMMA fragment helpers (layouts per CDNA5 ISA 7.12.2, bf16 16x16x32)
//  A (16x32 MxK): lane m = lane%16; half = lane/16
//     vec elems 0..7  = K kbase+8*half .. +7
//     vec elems 8..15 = K kbase+16+8*half .. +7
//  B (32x16 KxN): lane n = lane%16; same K order as A (elems 0..7 = kbase+8*half.., 8..15 = kbase+16+8*half..)
//  C/D (16x16):   d[r] -> (M = r + 8*half, N = lane%16)
// ---------------------------------------------------------------------------

static __device__ __forceinline__ v8f zero8() {
  v8f z;
#pragma unroll
  for (int i = 0; i < 8; ++i) z[i] = 0.0f;
  return z;
}

static __device__ __forceinline__ v16bf zero16bf() {
  v16bf z;
#pragma unroll
  for (int i = 0; i < 16; ++i) z[i] = (__bf16)0.0f;
  return z;
}

// A-fragment from a bf16 row (global or LDS): two contiguous 16B runs.
static __device__ __forceinline__ v16bf frag_a_bf(const __bf16* row, int kbase, int half) {
  const __bf16* p0 = row + kbase + 8 * half;
  const __bf16* p1 = p0 + 16;
  v16bf a;
#pragma unroll
  for (int e = 0; e < 8; ++e) { a[e] = p0[e]; a[e + 8] = p1[e]; }
  return a;
}

// A-fragment from an f32 row (only used for the raw input x).
static __device__ __forceinline__ v16bf frag_a_f32(const float* __restrict__ row, int kbase, int half) {
  const float* p0 = row + kbase + 8 * half;
  const float* p1 = p0 + 16;
  v16bf a;
#pragma unroll
  for (int e = 0; e < 8; ++e) { a[e] = (__bf16)p0[e]; a[e + 8] = (__bf16)p1[e]; }
  return a;
}

// B-fragment from a bf16 row: two contiguous 16B runs (same K order as A).
static __device__ __forceinline__ v16bf frag_b_bf(const __bf16* __restrict__ row, int kbase, int half) {
  const __bf16* p = row + kbase + 8 * half;
  v16bf b;
#pragma unroll
  for (int e = 0; e < 8; ++e) { b[e] = p[e]; b[e + 8] = p[e + 16]; }
  return b;
}

static __device__ __forceinline__ v8f wmma_bf16(v16bf a, v16bf b, v8f c) {
  // 8 args: (neg_a, A, neg_b, B, c_mod, C, reuse_a, reuse_b)
  return __builtin_amdgcn_wmma_f32_16x16x32_bf16(false, a, false, b, (short)0, c, false, false);
}

// ---------------------------------------------------------------------------
// Kernel 0a: rearrange 5x5 conv weights (O,I,5,5) f32 -> bf16 [tap][cout][cin]
// so each B-fragment is two contiguous 8-element runs over cin.
// ---------------------------------------------------------------------------
__global__ void k_wprep(const float* __restrict__ w_off1, __bf16* __restrict__ wB) {
  int i = blockIdx.x * 256 + threadIdx.x;           // total 25*64*64 = 102400
  if (i < 25 * NC * NC) {
    int co  = i / (NC * 25);
    int r   = i % (NC * 25);
    int ci  = r / 25;
    int tap = r % 25;
    wB[(tap * NC + co) * NC + ci] = (__bf16)w_off1[i];
  }
}

// Kernel 0b: elementwise f32 -> bf16 copy (1x1 weights; rows stay row-major,
// which is exactly the contiguous-in-K order the B-fragment wants).
__global__ void k_cvtw(const float* __restrict__ src, __bf16* __restrict__ dst, int n) {
  int i = blockIdx.x * 256 + threadIdx.x;
  if (i < n) dst[i] = (__bf16)src[i];
}

// ---------------------------------------------------------------------------
// Kernel 1: q = x @ wq^T + bq.  Stores q channels-last bf16 (conv input) and
// channels-first per-head bf16 (attention Q).
// ---------------------------------------------------------------------------
__global__ void k_qproj(const float* __restrict__ x, const __bf16* __restrict__ wq,
                        const float* __restrict__ bq,
                        __bf16* __restrict__ q_cl, __bf16* __restrict__ q_cf) {
  const int wave = threadIdx.x >> 5, lane = threadIdx.x & 31;
  const int nloc = lane & 15, half = lane >> 4;
  const int m0 = (blockIdx.x * 8 + wave) * 16;
  const float* arow = x + (m0 + nloc) * NC;
  v8f acc[4];
#pragma unroll
  for (int nt = 0; nt < 4; ++nt) acc[nt] = zero8();
#pragma unroll
  for (int ks = 0; ks < 2; ++ks) {
    v16bf a = frag_a_f32(arow, ks * 32, half);
#pragma unroll
    for (int nt = 0; nt < 4; ++nt) {
      v16bf b = frag_b_bf(wq + (nt * 16 + nloc) * NC, ks * 32, half);
      acc[nt] = wmma_bf16(a, b, acc[nt]);
    }
  }
#pragma unroll
  for (int nt = 0; nt < 4; ++nt) {
    const int n = nt * 16 + nloc;
    const float bias = bq[n];
#pragma unroll
    for (int r = 0; r < 8; ++r) {
      const int pixel = m0 + r + 8 * half;
      const __bf16 v = (__bf16)(acc[nt][r] + bias);
      q_cl[pixel * NC + n] = v;
      const int bb = pixel >> 14, ii = (pixel >> 7) & 127, dd = pixel & 127;
      q_cf[((bb * NC + n) << 14) + (ii << 7) + dd] = v;
    }
  }
}

// ---------------------------------------------------------------------------
// Kernel 2: 5x5 SAME conv (C->C) on q, im2col-on-the-fly.  200 WMMAs / tile.
// A and B fragments are pure bf16 b128 loads (no cvt in the hot loop).
// ---------------------------------------------------------------------------
__global__ void k_conv5(const __bf16* __restrict__ q_cl, const __bf16* __restrict__ wB,
                        const float* __restrict__ b_off1, float* __restrict__ cbuf) {
  const int wave = threadIdx.x >> 5, lane = threadIdx.x & 31;
  const int nloc = lane & 15, half = lane >> 4;
  const int m0 = (blockIdx.x * 8 + wave) * 16;      // tile = 16 pixels along W
  const int bb = m0 >> 14, hh = (m0 >> 7) & 127, w0 = m0 & 127;
  const int wm = w0 + nloc;                          // this lane's A-row column
  v8f acc[4];
#pragma unroll
  for (int nt = 0; nt < 4; ++nt) acc[nt] = zero8();
  for (int ty = 0; ty < 5; ++ty) {
    const int sr = hh + ty - 2;
    const bool rok = (unsigned)sr < (unsigned)HW;
    for (int tx = 0; tx < 5; ++tx) {
      const int sc = wm + tx - 2;
      const bool ok = rok && ((unsigned)sc < (unsigned)HW);
      const int aoff = ok ? ((bb * HW + sr) * HW + sc) * NC : 0;
      const int tap = ty * 5 + tx;
#pragma unroll
      for (int ks = 0; ks < 2; ++ks) {
        v16bf a = frag_a_bf(q_cl + aoff, ks * 32, half);
        if (!ok) a = zero16bf();                     // zero-pad rows (lane select)
#pragma unroll
        for (int nt = 0; nt < 4; ++nt) {
          v16bf b = frag_b_bf(wB + (tap * NC + nt * 16 + nloc) * NC, ks * 32, half);
          acc[nt] = wmma_bf16(a, b, acc[nt]);
        }
      }
    }
  }
#pragma unroll
  for (int nt = 0; nt < 4; ++nt) {
    const int n = nt * 16 + nloc;
    const float bias = b_off1[n];
#pragma unroll
    for (int r = 0; r < 8; ++r) {
      const int pixel = m0 + r + 8 * half;
      cbuf[pixel * NC + n] = acc[nt][r] + bias;      // f32: feeds tanh/grid math
    }
  }
}

// ---------------------------------------------------------------------------
// Kernel 3: offsets (1x1 to 2ch) -> tanh*5 -> grid -> bilinear sample of x.
// One wave per pixel; lanes split the 64 channels.  Sampled x stored bf16.
// ---------------------------------------------------------------------------
__global__ void k_offsample(const float* __restrict__ x, const float* __restrict__ cbuf,
                            const float* __restrict__ w_off2, __bf16* __restrict__ xs) {
  const int wave = threadIdx.x >> 5, lane = threadIdx.x & 31;
  const int p = blockIdx.x * 8 + wave;
  const float* crow = cbuf + (size_t)p * NC;
  const float c0 = crow[lane], c1 = crow[lane + 32];
  float s0 = c0 * w_off2[lane] + c1 * w_off2[lane + 32];
  float s1 = c0 * w_off2[NC + lane] + c1 * w_off2[NC + lane + 32];
#pragma unroll
  for (int m = 16; m >= 1; m >>= 1) {
    s0 += __shfl_xor(s0, m, 32);
    s1 += __shfl_xor(s1, m, 32);
  }
  const float off0 = tanhf(s0) * 5.0f;
  const float off1 = tanhf(s1) * 5.0f;
  const int bb = p >> 14, hh = (p >> 7) & 127, ww = p & 127;
  const float vx = (float)ww + off0;
  const float vy = (float)hh + off1;
  const float gx = 2.0f * vx / 127.0f - 1.0f;       // faithful: normalize by (H-1)
  const float gy = 2.0f * vy / 127.0f - 1.0f;
  const float ix = ((gx + 1.0f) * 128.0f - 1.0f) * 0.5f;
  const float iy = ((gy + 1.0f) * 128.0f - 1.0f) * 0.5f;
  const float ix0 = floorf(ix), iy0 = floorf(iy);
  const float fx = ix - ix0, fy = iy - iy0;
  const int jx0 = (int)ix0, jy0 = (int)iy0;
  const float wgt[4] = { (1.0f - fx) * (1.0f - fy), fx * (1.0f - fy),
                         (1.0f - fx) * fy,          fx * fy };
  const int cx[4] = { jx0, jx0 + 1, jx0, jx0 + 1 };
  const int cy[4] = { jy0, jy0, jy0 + 1, jy0 + 1 };
  float a0 = 0.0f, a1 = 0.0f;
#pragma unroll
  for (int k = 0; k < 4; ++k) {
    if ((unsigned)cx[k] < (unsigned)HW && (unsigned)cy[k] < (unsigned)HW) {
      const float* r = x + ((bb * HW + cy[k]) * HW + cx[k]) * NC;
      a0 += wgt[k] * r[lane];
      a1 += wgt[k] * r[lane + 32];
    }
  }
  xs[(size_t)p * NC + lane]      = (__bf16)a0;
  xs[(size_t)p * NC + lane + 32] = (__bf16)a1;
}

// ---------------------------------------------------------------------------
// Kernel 4: k,v projections (share A-fragments).
//   k stored bf16 (b,c,j,d) row-major   (QK^T B-operand: contiguous over d)
//   v stored bf16 (b,c,d,j) transposed  (P*V  B-operand: contiguous over j)
//   v += rel_bias[c*H + j]
// ---------------------------------------------------------------------------
__global__ void k_kvproj(const __bf16* __restrict__ xs,
                         const __bf16* __restrict__ wk, const float* __restrict__ bk,
                         const __bf16* __restrict__ wv, const float* __restrict__ bv,
                         const float* __restrict__ rel_bias,
                         __bf16* __restrict__ k_cf, __bf16* __restrict__ v_t) {
  const int wave = threadIdx.x >> 5, lane = threadIdx.x & 31;
  const int nloc = lane & 15, half = lane >> 4;
  const int m0 = (blockIdx.x * 8 + wave) * 16;
  const __bf16* arow = xs + (m0 + nloc) * NC;
  v8f ak[4], av[4];
#pragma unroll
  for (int nt = 0; nt < 4; ++nt) { ak[nt] = zero8(); av[nt] = zero8(); }
#pragma unroll
  for (int ks = 0; ks < 2; ++ks) {
    v16bf a = frag_a_bf(arow, ks * 32, half);
#pragma unroll
    for (int nt = 0; nt < 4; ++nt) {
      v16bf bkf = frag_b_bf(wk + (nt * 16 + nloc) * NC, ks * 32, half);
      ak[nt] = wmma_bf16(a, bkf, ak[nt]);
      v16bf bvf = frag_b_bf(wv + (nt * 16 + nloc) * NC, ks * 32, half);
      av[nt] = wmma_bf16(a, bvf, av[nt]);
    }
  }
#pragma unroll
  for (int nt = 0; nt < 4; ++nt) {
    const int n = nt * 16 + nloc;
    const float bkn = bk[n], bvn = bv[n];
#pragma unroll
    for (int r = 0; r < 8; ++r) {
      const int pixel = m0 + r + 8 * half;
      const int bb = pixel >> 14, jj = (pixel >> 7) & 127, dd = pixel & 127;
      const int head = bb * NC + n;
      k_cf[((size_t)head << 14) + (jj << 7) + dd] = (__bf16)(ak[nt][r] + bkn);
      v_t [((size_t)head << 14) + (dd << 7) + jj] =
          (__bf16)(av[nt][r] + bvn + rel_bias[(n << 7) + jj]);
    }
  }
}

// ---------------------------------------------------------------------------
// Kernel 5: per-head attention.  Block = 1 head (128x128), 8 waves, each wave
// owns 16 rows: S = QK^T (WMMA) -> row softmax (shfl) -> P staged bf16 in LDS
// -> O = P*V (WMMA, A-fragments from ds_load) -> obuf bf16 (b,c,i,d) flat.
// ---------------------------------------------------------------------------
__global__ void k_attn(const __bf16* __restrict__ q_cf, const __bf16* __restrict__ k_cf,
                       const __bf16* __restrict__ v_t, __bf16* __restrict__ obuf) {
  __shared__ __bf16 plds[8 * 16 * 128];              // 32 KB: P tiles, one per wave
  const int wave = threadIdx.x >> 5, lane = threadIdx.x & 31;
  const int nloc = lane & 15, half = lane >> 4;
  const int head = blockIdx.x;                       // 1024 heads
  const size_t hb = (size_t)head << 14;              // head * 128*128
  const int i0 = wave * 16;                          // this wave's row strip

  // ---- S = Q K^T -------------------------------------------------------
  v8f s[8];
#pragma unroll
  for (int jt = 0; jt < 8; ++jt) s[jt] = zero8();
  const __bf16* qrow = q_cf + hb + (size_t)(i0 + nloc) * 128;
#pragma unroll
  for (int ks = 0; ks < 4; ++ks) {
    v16bf a = frag_a_bf(qrow, ks * 32, half);
#pragma unroll
    for (int jt = 0; jt < 8; ++jt) {
      v16bf b = frag_b_bf(k_cf + hb + (size_t)(jt * 16 + nloc) * 128, ks * 32, half);
      s[jt] = wmma_bf16(a, b, s[jt]);
    }
  }

  // ---- row softmax (rows spread across 16 lanes of each half) ----------
  const float scale = 0.125f;                        // 64^-0.5
  __bf16* ptile = plds + wave * (16 * 128);
#pragma unroll
  for (int r = 0; r < 8; ++r) {
    float mx = -1.0e30f;
#pragma unroll
    for (int jt = 0; jt < 8; ++jt) mx = fmaxf(mx, s[jt][r]);
#pragma unroll
    for (int m = 8; m >= 1; m >>= 1) mx = fmaxf(mx, __shfl_xor(mx, m, 32));
    float e[8], sum = 0.0f;
#pragma unroll
    for (int jt = 0; jt < 8; ++jt) { e[jt] = __expf((s[jt][r] - mx) * scale); sum += e[jt]; }
#pragma unroll
    for (int m = 8; m >= 1; m >>= 1) sum += __shfl_xor(sum, m, 32);
    const float inv = 1.0f / sum;
    const int mrow = r + 8 * half;
#pragma unroll
    for (int jt = 0; jt < 8; ++jt)
      ptile[mrow * 128 + jt * 16 + nloc] = (__bf16)(e[jt] * inv);
  }
  __syncthreads();

  // ---- O = P V  (A from LDS bf16, B = v_t rows contiguous over j) ------
  v8f o[8];
#pragma unroll
  for (int dt = 0; dt < 8; ++dt) o[dt] = zero8();
  const __bf16* parow = ptile + nloc * 128;
#pragma unroll
  for (int ks = 0; ks < 4; ++ks) {
    v16bf a = frag_a_bf(parow, ks * 32, half);
#pragma unroll
    for (int dt = 0; dt < 8; ++dt) {
      v16bf b = frag_b_bf(v_t + hb + (size_t)(dt * 16 + nloc) * 128, ks * 32, half);
      o[dt] = wmma_bf16(a, b, o[dt]);
    }
  }
  __bf16* obase = obuf + hb;
#pragma unroll
  for (int dt = 0; dt < 8; ++dt) {
    const int dd = dt * 16 + nloc;
#pragma unroll
    for (int r = 0; r < 8; ++r) {
      const int ii = i0 + r + 8 * half;
      obase[ii * 128 + dd] = (__bf16)o[dt][r];
    }
  }
}

// ---------------------------------------------------------------------------
// Kernel 6: final projection.  obuf read FLAT -> the reference's raw reshape
// (b*c,h,w)->(b,h,w,c) is exactly this flat reinterpretation.  Output f32.
// ---------------------------------------------------------------------------
__global__ void k_outproj(const __bf16* __restrict__ obuf, const __bf16* __restrict__ w_out,
                          const float* __restrict__ b_out, float* __restrict__ y) {
  const int wave = threadIdx.x >> 5, lane = threadIdx.x & 31;
  const int nloc = lane & 15, half = lane >> 4;
  const int m0 = (blockIdx.x * 8 + wave) * 16;
  const __bf16* arow = obuf + (size_t)(m0 + nloc) * NC;
  v8f acc[4];
#pragma unroll
  for (int nt = 0; nt < 4; ++nt) acc[nt] = zero8();
#pragma unroll
  for (int ks = 0; ks < 2; ++ks) {
    v16bf a = frag_a_bf(arow, ks * 32, half);
#pragma unroll
    for (int nt = 0; nt < 4; ++nt) {
      v16bf b = frag_b_bf(w_out + (nt * 16 + nloc) * NC, ks * 32, half);
      acc[nt] = wmma_bf16(a, b, acc[nt]);
    }
  }
#pragma unroll
  for (int nt = 0; nt < 4; ++nt) {
    const int n = nt * 16 + nloc;
    const float bias = b_out[n];
#pragma unroll
    for (int r = 0; r < 8; ++r) {
      const int pixel = m0 + r + 8 * half;
      y[(size_t)pixel * NC + n] = acc[nt][r] + bias;
    }
  }
}

// ---------------------------------------------------------------------------
extern "C" void kernel_launch(void* const* d_in, const int* in_sizes, int n_in,
                              void* d_out, int out_size, void* d_ws, size_t ws_size,
                              hipStream_t stream) {
  (void)in_sizes; (void)n_in; (void)out_size; (void)ws_size;
  const float* x        = (const float*)d_in[0];
  const float* wq       = (const float*)d_in[1];
  const float* bq       = (const float*)d_in[2];
  const float* wk       = (const float*)d_in[3];
  const float* bk       = (const float*)d_in[4];
  const float* wv       = (const float*)d_in[5];
  const float* bv       = (const float*)d_in[6];
  const float* w_off1   = (const float*)d_in[7];
  const float* b_off1   = (const float*)d_in[8];
  const float* w_off2   = (const float*)d_in[9];
  const float* w_out    = (const float*)d_in[10];
  const float* b_out    = (const float*)d_in[11];
  const float* rel_bias = (const float*)d_in[12];
  float* y = (float*)d_out;

  // Workspace: 4 x 64MB regions (bf16 tensors under-fill them; aliasing across
  // stage lifetimes unchanged) + bf16 weight buffers.
  float* ws      = (float*)d_ws;
  __bf16* q_cl   = (__bf16*)(ws + 0 * NBUF);  // dead after conv5    -> reused as v_t
  __bf16* q_cf   = (__bf16*)(ws + 1 * NBUF);  // live until attn
  float*  cbuf   = (float*) (ws + 2 * NBUF);  // f32; dead after offsample -> reused as k_cf
  __bf16* xs     = (__bf16*)(ws + 3 * NBUF);  // dead after kvproj   -> reused as obuf
  __bf16* wB     = (__bf16*)(ws + 4 * NBUF);            // 25*64*64 bf16
  __bf16* wq_bf  = wB + 25 * NC * NC;
  __bf16* wk_bf  = wq_bf + NC * NC;
  __bf16* wv_bf  = wk_bf + NC * NC;
  __bf16* wo_bf  = wv_bf + NC * NC;
  __bf16* k_cf   = (__bf16*)cbuf;
  __bf16* v_t    = q_cl;
  __bf16* obuf   = xs;

  const int gemm_blocks = NPIX / (8 * 16);           // 2048 (8 waves x 16-pixel tiles)

  k_wprep    <<<400,         256, 0, stream>>>(w_off1, wB);
  k_cvtw     <<<16,          256, 0, stream>>>(wq,    wq_bf, NC * NC);
  k_cvtw     <<<16,          256, 0, stream>>>(wk,    wk_bf, NC * NC);
  k_cvtw     <<<16,          256, 0, stream>>>(wv,    wv_bf, NC * NC);
  k_cvtw     <<<16,          256, 0, stream>>>(w_out, wo_bf, NC * NC);
  k_qproj    <<<gemm_blocks, 256, 0, stream>>>(x, wq_bf, bq, q_cl, q_cf);
  k_conv5    <<<gemm_blocks, 256, 0, stream>>>(q_cl, wB, b_off1, cbuf);
  k_offsample<<<NPIX / 8,    256, 0, stream>>>(x, cbuf, w_off2, xs);
  k_kvproj   <<<gemm_blocks, 256, 0, stream>>>(xs, wk_bf, bk, wv_bf, bv, rel_bias, k_cf, v_t);
  k_attn     <<<NBATCH * NC, 256, 0, stream>>>(q_cf, k_cf, v_t, obuf);
  k_outproj  <<<gemm_blocks, 256, 0, stream>>>(obuf, w_out ? wo_bf : wo_bf, b_out, y);
}
